// MultiHeadSelfAttention_70085276336856
// MI455X (gfx1250) — matrix operand order repaired and checked by reading, not compiled, not run on hardware
//
#include <hip/hip_runtime.h>
#include <hip/hip_bf16.h>

// ---------------------------------------------------------------------------
// MI455X (gfx1250) fused multi-head self-attention, bf16 WMMA path.
//   B=4, S=2048, D=1024, H=16, dk=dv=64.
// Pipeline: fp32->bf16 convert  ->  QKV GEMMs (bf16 WMMA, f32 acc, async LDS
//           staging, double buffered) -> flash attention -> output GEMM (f32).
// ---------------------------------------------------------------------------

typedef __attribute__((ext_vector_type(16))) __bf16 v16bf;
typedef __attribute__((ext_vector_type(8)))  float  v8f;
typedef __attribute__((ext_vector_type(4)))  unsigned short us4;

union FragB16 {            // one WMMA 16-bit operand fragment (8 VGPRs / lane)
    v16bf v;
    uint4 q[2];
    unsigned short s[16];
};

__device__ __forceinline__ unsigned short f2bf(float f) {
    union { float f; unsigned u; } x; x.f = f;
    unsigned r = x.u + 0x7FFFu + ((x.u >> 16) & 1u);   // round-to-nearest-even
    return (unsigned short)(r >> 16);
}

__device__ __forceinline__ v8f wmma_bf16(v16bf a, v16bf b, v8f c) {
    // v_wmma_f32_16x16x32_bf16  (neg_a, A, neg_b, B, c_mod, C, reuse_a, reuse_b)
    return __builtin_amdgcn_wmma_f32_16x16x32_bf16(false, a, false, b, (short)0, c,
                                                   false, false);
}

// Per-lane async DMA: 16 bytes global -> LDS, tracked by ASYNCcnt, no VGPR data.
__device__ __forceinline__ void async_load_b128(const unsigned short* g,
                                                unsigned short* l) {
    unsigned lds_off = (unsigned)(size_t)l;   // low 32 bits = LDS byte offset
    asm volatile("global_load_async_to_lds_b128 %0, %1, off"
                 :: "v"(lds_off), "v"(g) : "memory");
}
__device__ __forceinline__ void wait_asynccnt0() {
    asm volatile("s_wait_asynccnt 0" ::: "memory");
}

// ---------------------------------------------------------------------------
// fp32 -> bf16 convert (vectorized x4)
// ---------------------------------------------------------------------------
__global__ void cvt_f32_to_bf16(const float* __restrict__ in,
                                unsigned short* __restrict__ out, int n4) {
    int i = blockIdx.x * blockDim.x + threadIdx.x;
    int stride = gridDim.x * blockDim.x;
    for (int j = i; j < n4; j += stride) {
        float4 f = ((const float4*)in)[j];
        us4 o;
        o.x = f2bf(f.x); o.y = f2bf(f.y); o.z = f2bf(f.z); o.w = f2bf(f.w);
        ((us4*)out)[j] = o;
    }
}

// ---------------------------------------------------------------------------
// C[M,N] = A[M,K] * W[N,K]^T   (bf16 inputs, f32 accumulate)
// Block tile 128x128, BK=64, 256 threads (8 waves); wave computes 64x32.
// Double-buffered LDS staging via global_load_async_to_lds_b128.
// OUT_BF16: true -> bf16 output (QKV), false -> f32 output (final proj).
// ---------------------------------------------------------------------------
constexpr int GLDT = 72;   // 64 + 8 bf16 pad (keeps 16B alignment: 72*2=144)

__device__ __forceinline__ void gemm_stage_async(
    const unsigned short* __restrict__ A, const unsigned short* __restrict__ W,
    unsigned short* sa, unsigned short* sb,
    int tid, int m0, int n0, int k0, int Kdim) {
#pragma unroll
    for (int i = 0; i < 4; ++i) {
        int c   = tid + 256 * i;      // 0..1023 chunk id
        int row = c >> 3;
        int col = c & 7;
        async_load_b128(&A[(size_t)(m0 + row) * Kdim + k0 + col * 8],
                        sa + row * GLDT + col * 8);
        async_load_b128(&W[(size_t)(n0 + row) * Kdim + k0 + col * 8],
                        sb + row * GLDT + col * 8);
    }
}

template <bool OUT_BF16>
__global__ __launch_bounds__(256) void gemm_bf16_wmma(
    const unsigned short* __restrict__ A,
    const unsigned short* __restrict__ W,
    void* __restrict__ Cout, int Mdim, int Ndim, int Kdim)
{
    constexpr int BM = 128, BK = 64, LDT = GLDT;
    __shared__ alignas(16) unsigned short sA[2][BM * LDT];
    __shared__ alignas(16) unsigned short sB[2][BM * LDT];

    const int tid   = threadIdx.x;
    const int lane  = tid & 31;
    const int wave  = tid >> 5;       // 0..7
    const int wm    = wave >> 2;      // 0..1  (64-row strip)
    const int wn    = wave & 3;       // 0..3  (32-col strip)
    const int lhalf = lane >> 4;      // 0/1
    const int l16   = lane & 15;

    const int m0 = blockIdx.y * BM;
    const int n0 = blockIdx.x * BM;

    v8f acc[4][2];
#pragma unroll
    for (int mi = 0; mi < 4; ++mi)
#pragma unroll
        for (int ni = 0; ni < 2; ++ni)
#pragma unroll
            for (int e = 0; e < 8; ++e) acc[mi][ni][e] = 0.0f;

    // Prologue: fill buffer 0 asynchronously.
    gemm_stage_async(A, W, sA[0], sB[0], tid, m0, n0, 0, Kdim);

    int buf = 0;
    for (int k0 = 0; k0 < Kdim; k0 += BK) {
        wait_asynccnt0();             // current buffer filled (this wave's DMAs)
        __syncthreads();              // everyone filled + prior reads retired
        if (k0 + BK < Kdim)           // prefetch next tile into other buffer
            gemm_stage_async(A, W, sA[buf ^ 1], sB[buf ^ 1], tid, m0, n0,
                             k0 + BK, Kdim);

        const unsigned short* la = sA[buf];
        const unsigned short* lb = sB[buf];
#pragma unroll
        for (int ks = 0; ks < 2; ++ks) {          // two K=32 steps per tile
            FragB16 af[4], bf[2];
#pragma unroll
            for (int mi = 0; mi < 4; ++mi) {      // A frag: row m, k split 8+8
                int r  = wm * 64 + mi * 16 + l16;
                int kb = ks * 32 + lhalf * 8;
                af[mi].q[0] = *(const uint4*)&la[r * LDT + kb];
                af[mi].q[1] = *(const uint4*)&la[r * LDT + kb + 16];
            }
#pragma unroll
            for (int ni = 0; ni < 2; ++ni) {      // B frag: col n, k split 8+8
                int r  = wn * 32 + ni * 16 + l16;
                int kb = ks * 32 + lhalf * 8;
                bf[ni].q[0] = *(const uint4*)&lb[r * LDT + kb];
                bf[ni].q[1] = *(const uint4*)&lb[r * LDT + kb + 16];
            }
#pragma unroll
            for (int mi = 0; mi < 4; ++mi)
#pragma unroll
                for (int ni = 0; ni < 2; ++ni)
                    acc[mi][ni] = wmma_bf16(af[mi].v, bf[ni].v, acc[mi][ni]);
        }
        buf ^= 1;
    }

    // Writeback: C layout elem j -> row = j + 8*lhalf, col = l16.
#pragma unroll
    for (int mi = 0; mi < 4; ++mi)
#pragma unroll
        for (int ni = 0; ni < 2; ++ni) {
            int col = n0 + wn * 32 + ni * 16 + l16;
#pragma unroll
            for (int j = 0; j < 8; ++j) {
                int row = m0 + wm * 64 + mi * 16 + j + 8 * lhalf;
                float v = acc[mi][ni][j];
                if (OUT_BF16)
                    ((unsigned short*)Cout)[(size_t)row * Ndim + col] = f2bf(v);
                else
                    ((float*)Cout)[(size_t)row * Ndim + col] = v;
            }
        }
}

// ---------------------------------------------------------------------------
// Causal flash attention.  One WG per (b, h, 64-row q-tile); 4 waves, each
// owning 16 q-rows.  bf16 WMMA for QK^T and PV, f32 online softmax.
// K tile staged via async-to-LDS DMA; V staged transposed manually.
// ---------------------------------------------------------------------------
__global__ __launch_bounds__(128) void flash_attn_bf16(
    const unsigned short* __restrict__ Q,
    const unsigned short* __restrict__ K,
    const unsigned short* __restrict__ V,
    unsigned short* __restrict__ O)
{
    constexpr int S = 2048, D = 1024, DK = 64, LDT = DK + 8;
    const float scale = 0.125f;                       // 1/sqrt(64)

    __shared__ alignas(16) unsigned short sK[64 * LDT];   // K tile [n][d]
    __shared__ alignas(16) unsigned short sVt[64 * LDT];  // V^T tile [dv][n]
    __shared__ alignas(16) unsigned short sP[4][16 * LDT];// per-wave probs

    const int tid   = threadIdx.x;
    const int lane  = tid & 31;
    const int wave  = tid >> 5;
    const int lhalf = lane >> 4;
    const int l16   = lane & 15;

    const int qt = blockIdx.x;                        // 0..31
    const int h  = blockIdx.y;                        // 0..15
    const int b  = blockIdx.z;                        // 0..3
    const size_t headoff = (size_t)h * DK;
    const size_t baseBS  = (size_t)b * S;

    // Q A-fragments (held for whole kernel): lane row = q0+wave*16+l16.
    const int qrow = qt * 64 + wave * 16 + l16;
    FragB16 qf[2];
    {
        const unsigned short* qp = Q + (baseBS + qrow) * D + headoff;
#pragma unroll
        for (int ks = 0; ks < 2; ++ks) {
            qf[ks].q[0] = *(const uint4*)(qp + ks * 32 + lhalf * 8);
            qf[ks].q[1] = *(const uint4*)(qp + ks * 32 + 16 + lhalf * 8);
        }
    }

    v8f oacc[4];
#pragma unroll
    for (int t = 0; t < 4; ++t)
#pragma unroll
        for (int e = 0; e < 8; ++e) oacc[t][e] = 0.0f;
    float rm[8], rl[8];
#pragma unroll
    for (int j = 0; j < 8; ++j) { rm[j] = -1e30f; rl[j] = 0.0f; }

    const int nkt = qt + 1;                           // causal: tiles 0..qt
    for (int kt = 0; kt < nkt; ++kt) {
        const int kv0 = kt * 64;
        __syncthreads();                  // prior tile fully consumed
        // Stage K tile via async DMA: 512 x 16B chunks, 4 per thread.
#pragma unroll
        for (int i = 0; i < 4; ++i) {
            int c = tid + 128 * i;
            int row = c >> 3, col = c & 7;
            async_load_b128(&K[(baseBS + kv0 + row) * D + headoff + col * 8],
                            &sK[row * LDT + col * 8]);
        }
        // Stage V tile transposed: thread -> (row, 32-wide dv half).
        {
            int row = tid >> 1, half = tid & 1;
            union { uint4 q[4]; unsigned short s[32]; } vv;
            const unsigned short* vp =
                &V[(baseBS + kv0 + row) * D + headoff + half * 32];
            vv.q[0] = *(const uint4*)(vp);
            vv.q[1] = *(const uint4*)(vp + 8);
            vv.q[2] = *(const uint4*)(vp + 16);
            vv.q[3] = *(const uint4*)(vp + 24);
#pragma unroll
            for (int i = 0; i < 32; ++i)
                sVt[(half * 32 + i) * LDT + row] = vv.s[i];
        }
        wait_asynccnt0();                 // K DMA landed in LDS
        __syncthreads();                  // (compiler also waits DS stores)

        // S = (Q K^T) * scale, 16x64 strip per wave (4 n-tiles).
        v8f sc[4];
#pragma unroll
        for (int ni = 0; ni < 4; ++ni) {
#pragma unroll
            for (int e = 0; e < 8; ++e) sc[ni][e] = 0.0f;
#pragma unroll
            for (int ks = 0; ks < 2; ++ks) {
                FragB16 kf;
                int r  = ni * 16 + l16;
                int kb = ks * 32 + lhalf * 8;
                kf.q[0] = *(const uint4*)&sK[r * LDT + kb];
                kf.q[1] = *(const uint4*)&sK[r * LDT + kb + 16];
                sc[ni] = wmma_bf16(qf[ks].v, kf.v, sc[ni]);
            }
#pragma unroll
            for (int j = 0; j < 8; ++j) sc[ni][j] *= scale;
        }

        // Causal mask (only the diagonal tile can be partially masked).
        if (kt == qt) {
#pragma unroll
            for (int ni = 0; ni < 4; ++ni) {
                int kcol = kv0 + ni * 16 + l16;
#pragma unroll
                for (int j = 0; j < 8; ++j) {
                    int qr = qt * 64 + wave * 16 + j + 8 * lhalf;
                    if (kcol > qr) sc[ni][j] = -1e30f;
                }
            }
        }

        // Online softmax: row stats live in the 16 lanes of one half-wave.
        float tmax[8];
#pragma unroll
        for (int j = 0; j < 8; ++j)
            tmax[j] = fmaxf(fmaxf(sc[0][j], sc[1][j]),
                            fmaxf(sc[2][j], sc[3][j]));
#pragma unroll
        for (int j = 0; j < 8; ++j)
            for (int off = 1; off < 16; off <<= 1)
                tmax[j] = fmaxf(tmax[j], __shfl_xor(tmax[j], off, 32));

        float al[8];
#pragma unroll
        for (int j = 0; j < 8; ++j) {
            float nm = fmaxf(rm[j], tmax[j]);
            al[j]  = __expf(rm[j] - nm);
            rm[j]  = nm;
        }
        float tsum[8];
#pragma unroll
        for (int j = 0; j < 8; ++j) tsum[j] = 0.0f;
#pragma unroll
        for (int ni = 0; ni < 4; ++ni)
#pragma unroll
            for (int j = 0; j < 8; ++j) {
                float p = __expf(sc[ni][j] - rm[j]);
                sc[ni][j] = p;
                tsum[j] += p;
            }
#pragma unroll
        for (int j = 0; j < 8; ++j)
            for (int off = 1; off < 16; off <<= 1)
                tsum[j] += __shfl_xor(tsum[j], off, 32);
#pragma unroll
        for (int j = 0; j < 8; ++j) rl[j] = rl[j] * al[j] + tsum[j];
#pragma unroll
        for (int t = 0; t < 4; ++t)
#pragma unroll
            for (int j = 0; j < 8; ++j) oacc[t][j] *= al[j];

        // P (C-layout f32) -> per-wave LDS bf16 -> A-layout fragments.
#pragma unroll
        for (int ni = 0; ni < 4; ++ni)
#pragma unroll
            for (int j = 0; j < 8; ++j)
                sP[wave][(j + 8 * lhalf) * LDT + ni * 16 + l16] =
                    f2bf(sc[ni][j]);
        __asm__ volatile("s_wait_dscnt 0" ::: "memory");

        FragB16 pf[2];
#pragma unroll
        for (int ks = 0; ks < 2; ++ks) {
            int kb = ks * 32 + lhalf * 8;
            pf[ks].q[0] = *(const uint4*)&sP[wave][l16 * LDT + kb];
            pf[ks].q[1] = *(const uint4*)&sP[wave][l16 * LDT + kb + 16];
        }
        // O += P * V   (V^T staged so B-fragments are contiguous rows).
#pragma unroll
        for (int t = 0; t < 4; ++t)
#pragma unroll
            for (int ks = 0; ks < 2; ++ks) {
                FragB16 vf;
                int r  = t * 16 + l16;            // dv row of sVt
                int kb = ks * 32 + lhalf * 8;     // k = n within tile
                vf.q[0] = *(const uint4*)&sVt[r * LDT + kb];
                vf.q[1] = *(const uint4*)&sVt[r * LDT + kb + 16];
                oacc[t] = wmma_bf16(pf[ks].v, vf.v, oacc[t]);
            }
    }

    // Normalize and store concat layout [b, s, h*64 + dv] as bf16.
    float rr[8];
#pragma unroll
    for (int j = 0; j < 8; ++j) rr[j] = 1.0f / rl[j];
#pragma unroll
    for (int t = 0; t < 4; ++t) {
        int col = t * 16 + l16;
#pragma unroll
        for (int j = 0; j < 8; ++j) {
            int qr = qt * 64 + wave * 16 + j + 8 * lhalf;
            O[(baseBS + qr) * D + headoff + col] = f2bf(oacc[t][j] * rr[j]);
        }
    }
}

// ---------------------------------------------------------------------------
// Host-side launch
// ---------------------------------------------------------------------------
extern "C" void kernel_launch(void* const* d_in, const int* in_sizes, int n_in,
                              void* d_out, int out_size, void* d_ws, size_t ws_size,
                              hipStream_t stream) {
    (void)in_sizes; (void)n_in; (void)out_size; (void)ws_size;

    constexpr int Bc = 4, Sc = 2048, Dc = 1024, Hc = 16;
    constexpr int Mrows = Bc * Sc;                       // 8192

    const float* Wq = (const float*)d_in[1];
    const float* Wk = (const float*)d_in[2];
    const float* Wv = (const float*)d_in[3];
    const float* Wo = (const float*)d_in[4];
    const float* X  = (const float*)d_in[5];

    const size_t nX = (size_t)Mrows * Dc;                // 8,388,608
    const size_t nW = (size_t)Dc * Dc;                   // 1,048,576

    unsigned short* ws  = (unsigned short*)d_ws;         // ~88 MB total
    unsigned short* Xb  = ws;
    unsigned short* Wqb = Xb  + nX;
    unsigned short* Wkb = Wqb + nW;
    unsigned short* Wvb = Wkb + nW;
    unsigned short* Wob = Wvb + nW;
    unsigned short* Qb  = Wob + nW;
    unsigned short* Kb  = Qb  + nX;
    unsigned short* Vb  = Kb  + nX;
    unsigned short* Ab  = Vb  + nX;

    // 1) fp32 -> bf16 converts
    {
        int n4 = (int)(nX / 4);
        cvt_f32_to_bf16<<<dim3((n4 + 255) / 256), dim3(256), 0, stream>>>(X, Xb, n4);
        int w4 = (int)(nW / 4);
        dim3 gw((w4 + 255) / 256);
        cvt_f32_to_bf16<<<gw, dim3(256), 0, stream>>>(Wq, Wqb, w4);
        cvt_f32_to_bf16<<<gw, dim3(256), 0, stream>>>(Wk, Wkb, w4);
        cvt_f32_to_bf16<<<gw, dim3(256), 0, stream>>>(Wv, Wvb, w4);
        cvt_f32_to_bf16<<<gw, dim3(256), 0, stream>>>(Wo, Wob, w4);
    }

    // 2) QKV projections: [8192,1024] x [1024,1024]^T, bf16 out
    dim3 gg(Dc / 128, Mrows / 128), gb(256);
    gemm_bf16_wmma<true><<<gg, gb, 0, stream>>>(Xb, Wqb, Qb, Mrows, Dc, Dc);
    gemm_bf16_wmma<true><<<gg, gb, 0, stream>>>(Xb, Wkb, Kb, Mrows, Dc, Dc);
    gemm_bf16_wmma<true><<<gg, gb, 0, stream>>>(Xb, Wvb, Vb, Mrows, Dc, Dc);

    // 3) causal flash attention -> concat bf16
    flash_attn_bf16<<<dim3(Sc / 64, Hc, Bc), dim3(128), 0, stream>>>(Qb, Kb, Vb, Ab);

    // 4) output projection, f32 out
    gemm_bf16_wmma<false><<<gg, gb, 0, stream>>>(Ab, Wob, d_out, Mrows, Dc, Dc);
}
